// DLRM_23922967838956
// MI455X (gfx1250) — matrix operand order repaired and checked by reading, not compiled, not run on hardware
//
#include <hip/hip_runtime.h>
#include <stdint.h>

typedef __bf16 v16bf __attribute__((ext_vector_type(16)));
typedef float  v8f   __attribute__((ext_vector_type(8)));

#define BATCH 16384
#define NSPARSE 26
#define VOCAB 100000
#define DIM 128

// ---------------------------------------------------------------------------
// Weight conversion with transpose:  src f32 [K][N]  ->  dst bf16 [N][Kp]
// (pad rows k >= K are zero so padded-K activation garbage cannot leak)
// ---------------------------------------------------------------------------
__global__ __launch_bounds__(256) void convert_pad_t(const float* __restrict__ src,
                                                     __bf16* __restrict__ dst,
                                                     int K, int N, int Kp) {
    int i = blockIdx.x * 256 + threadIdx.x;      // over N*Kp
    if (i >= N * Kp) return;
    int n = i / Kp, k = i - n * Kp;
    dst[i] = (k < K) ? (__bf16)src[(size_t)k * N + n] : (__bf16)0.0f;
}

__global__ __launch_bounds__(256) void pad_dense(const float* __restrict__ src,
                                                 __bf16* __restrict__ dst) {
    int i = blockIdx.x * 256 + threadIdx.x;      // over BATCH*32
    if (i >= BATCH * 32) return;
    int b = i >> 5, c = i & 31;
    dst[i] = (c < 13) ? (__bf16)src[b * 13 + c] : (__bf16)0.0f;
}

// ---------------------------------------------------------------------------
// CDNA5 async copy: 16 bytes per lane, global -> LDS, tracked by ASYNCcnt
// ---------------------------------------------------------------------------
__device__ __forceinline__ void async_b128(uint32_t lds_byte_off, const void* gaddr) {
    asm volatile("global_load_async_to_lds_b128 %0, %1, off"
                 :: "v"(lds_byte_off), "v"(gaddr) : "memory");
}

// ---------------------------------------------------------------------------
// Tiled bf16 GEMM:  C[M,N] = bf16( relu( A[M,K] * Wt[N,K]^T + bias[N] ) )
// Wt is pre-transposed [N][K].  256 threads (8 waves); block tile 128x128;
// wave tile 32(M) x 64(N); K-step 32; double-buffered LDS fed by
// global_load_async_to_lds_b128; 8 back-to-back WMMAs per wave per K-step.
// ---------------------------------------------------------------------------
__global__ __launch_bounds__(256)
void gemm_bf16_relu(const __bf16* __restrict__ A, const __bf16* __restrict__ Wt,
                    const float* __restrict__ bias, __bf16* __restrict__ C,
                    int M, int N, int K) {
    __shared__ __align__(16) __bf16 As[2][128 * 32];   // [m][k]   2 x 8KB
    __shared__ __align__(16) __bf16 Bs[2][128 * 32];   // [n][k]   2 x 8KB

    const int t     = threadIdx.x;
    const int lane  = t & 31;
    const int wave  = t >> 5;
    const int hs    = lane >> 4;          // lane-half select
    const int lm    = lane & 15;
    const int waveM = (wave & 3) * 32;
    const int waveN = (wave >> 2) * 64;
    const int tileM = (int)blockIdx.y * 128;
    const int tileN = (int)blockIdx.x * 128;

    const uint32_t asBase = (uint32_t)(size_t)(&As[0][0]);
    const uint32_t bsBase = (uint32_t)(size_t)(&Bs[0][0]);

    v8f acc[2][4];
    const v8f vzero = {0.f,0.f,0.f,0.f,0.f,0.f,0.f,0.f};
#pragma unroll
    for (int mt = 0; mt < 2; mt++)
#pragma unroll
        for (int nt = 0; nt < 4; nt++) acc[mt][nt] = vzero;

    // per-thread tile-copy coordinates (both tiles are plain 128x32 copies)
    const int q0 = t;                  // uint4 index 0..255
    const int q1 = 256 + t;            // uint4 index 256..511
    const int r0 = q0 >> 2, c0 = (q0 & 3) * 8;   // row, bf16 col
    const int r1 = q1 >> 2, c1 = (q1 & 3) * 8;

    const __bf16* Abase = A  + (size_t)tileM * K;
    const __bf16* Bbase = Wt + (size_t)tileN * K;

    auto issue_tile = [&](int k0, int buf) {
        uint32_t ao = asBase + (uint32_t)buf * 8192;
        uint32_t bo = bsBase + (uint32_t)buf * 8192;
        async_b128(ao + q0 * 16, Abase + (size_t)r0 * K + k0 + c0);
        async_b128(ao + q1 * 16, Abase + (size_t)r1 * K + k0 + c1);
        async_b128(bo + q0 * 16, Bbase + (size_t)r0 * K + k0 + c0);
        async_b128(bo + q1 * 16, Bbase + (size_t)r1 * K + k0 + c1);
    };

    const int S = K >> 5;              // number of K-steps
    issue_tile(0, 0);

    for (int i = 0; i < S; i++) {
        const int buf = i & 1;
        if (i + 1 < S) {
            issue_tile((i + 1) << 5, buf ^ 1);
            // 4 newer async ops may remain in flight; ours (older 4) are done
            asm volatile("s_wait_asynccnt 0x4" ::: "memory");
        } else {
            asm volatile("s_wait_asynccnt 0x0" ::: "memory");
        }
        __syncthreads();

        // ---- preload ALL fragments, then 8 back-to-back WMMAs ----
        union F { v16bf v; uint4 q[2]; };
        F af[2], bfr[4];
#pragma unroll
        for (int mt = 0; mt < 2; mt++) {
            const uint4* Ar = (const uint4*)(&As[buf][0] + (waveM + mt * 16 + lm) * 32);
            af[mt].q[0] = Ar[hs];          // dwords hs*4   (K 0..7 / 8..15)
            af[mt].q[1] = Ar[2 + hs];      // dwords 8+hs*4 (K 16..23 / 24..31)
        }
#pragma unroll
        for (int nt = 0; nt < 4; nt++) {
            const uint4* Br = (const uint4*)(&Bs[buf][0] + (waveN + nt * 16 + lm) * 32);
            bfr[nt].q[0] = Br[hs];
            bfr[nt].q[1] = Br[2 + hs];
        }
#pragma unroll
        for (int nt = 0; nt < 4; nt++)
#pragma unroll
            for (int mt = 0; mt < 2; mt++)
                acc[mt][nt] = __builtin_amdgcn_wmma_f32_16x16x32_bf16(
                    false, af[mt].v, false, bfr[nt].v, (short)0, acc[mt][nt], false, false);
        __syncthreads();   // reads done before this buffer is overwritten (i+2)
    }

    // ---- epilogue: bias + relu + bf16 store ----
#pragma unroll
    for (int mt = 0; mt < 2; mt++) {
        const int mbase = tileM + waveM + mt * 16 + hs * 8;
#pragma unroll
        for (int nt = 0; nt < 4; nt++) {
            int n = tileN + waveN + nt * 16 + lm;
            float bv = bias[n];
#pragma unroll
            for (int r = 0; r < 8; r++) {
                float v = acc[mt][nt][r] + bv;
                v = v > 0.f ? v : 0.f;
                C[(size_t)(mbase + r) * N + n] = (__bf16)v;
            }
        }
    }
}

// ---------------------------------------------------------------------------
// Gather: Feat[b][0][:] = dense_out; Feat[b][1+f][:] = emb[f][sparse[b][f]][:]
// also X0[b][0:128] = dense_out, X0[b][479:512] = 0 (NaN guard on padded K)
// ---------------------------------------------------------------------------
__global__ __launch_bounds__(128)
void gather_feats(const float* __restrict__ emb, const int* __restrict__ sparse,
                  const __bf16* __restrict__ Dout, __bf16* __restrict__ Feat,
                  __bf16* __restrict__ X0) {
    const int b = blockIdx.x;
    const int t = threadIdx.x;              // 0..127
    __bf16* Fb = Feat + (size_t)b * 32 * DIM;
    __bf16 dv = Dout[(size_t)b * DIM + t];
    Fb[t] = dv;
    X0[(size_t)b * 512 + t] = dv;
    if (t < 33) X0[(size_t)b * 512 + 479 + t] = (__bf16)0.0f;
#pragma unroll 1
    for (int f = 0; f < NSPARSE; f++) {
        int idx = sparse[b * NSPARSE + f];
        float v = emb[((size_t)f * VOCAB + idx) * DIM + t];
        Fb[(f + 1) * DIM + t] = (__bf16)v;
    }
}

// ---------------------------------------------------------------------------
// Interaction: per sample G = F * F^T (F = 32x128 bf16, rows 27..31 unused),
// scatter triu(27, k=1) -> X0[b][128 + idx].  One wave per sample.
// ---------------------------------------------------------------------------
__global__ __launch_bounds__(128)
void interact(const __bf16* __restrict__ Feat, __bf16* __restrict__ X0) {
    const int lane = threadIdx.x & 31;
    const int wave = threadIdx.x >> 5;
    const int b    = (int)blockIdx.x * 4 + wave;
    const uint32_t* F = (const uint32_t*)(Feat + (size_t)b * 32 * DIM); // 64 dwords/row
    const int hs = lane >> 4;
    const int lm = lane & 15;

    const v8f vzero = {0.f,0.f,0.f,0.f,0.f,0.f,0.f,0.f};
    v8f a00 = vzero, a01 = vzero, a11 = vzero;

    for (int k0 = 0; k0 < 64; k0 += 16) {   // dword units: K=128 bf16 total
        union { v16bf v; uint32_t u[8]; } A0, A1, B0, B1;
#pragma unroll
        for (int i = 0; i < 8; i++) {
            int dj = k0 + (i & 3) + ((i >> 2) << 3) + (hs << 2);   // A layout
            A0.u[i] = F[(size_t)(lm)      * 64 + dj];
            A1.u[i] = F[(size_t)(16 + lm) * 64 + dj];
            int dk = k0 + (i & 3) + ((i >> 2) << 3) + (hs << 2);   // B layout
            B0.u[i] = F[(size_t)(lm)      * 64 + dk];
            B1.u[i] = F[(size_t)(16 + lm) * 64 + dk];
        }
        a00 = __builtin_amdgcn_wmma_f32_16x16x32_bf16(false, A0.v, false, B0.v, (short)0, a00, false, false);
        a01 = __builtin_amdgcn_wmma_f32_16x16x32_bf16(false, A0.v, false, B1.v, (short)0, a01, false, false);
        a11 = __builtin_amdgcn_wmma_f32_16x16x32_bf16(false, A1.v, false, B1.v, (short)0, a11, false, false);
    }

    __bf16* xr = X0 + (size_t)b * 512 + 128;
    auto emit = [&](v8f acc, int mt, int nt) {
        int j = nt * 16 + lm;
#pragma unroll
        for (int r = 0; r < 8; r++) {
            int i = mt * 16 + r + hs * 8;
            if (i < j && j <= 26) {
                int idx = 26 * i - (i * (i - 1)) / 2 + (j - i - 1);
                xr[idx] = (__bf16)acc[r];
            }
        }
    };
    emit(a00, 0, 0);
    emit(a01, 0, 1);
    emit(a11, 1, 1);
}

// ---------------------------------------------------------------------------
// Final layer: out[b] = relu( X4[b,:256] . w4 + b4 )
// ---------------------------------------------------------------------------
__global__ __launch_bounds__(256)
void final_dot(const __bf16* __restrict__ X4, const float* __restrict__ w4,
               const float* __restrict__ b4, float* __restrict__ out) {
    int b = blockIdx.x * 256 + threadIdx.x;
    if (b >= BATCH) return;
    float acc = b4[0];
    const __bf16* x = X4 + (size_t)b * 256;
#pragma unroll 4
    for (int k = 0; k < 256; k++) acc += (float)x[k] * w4[k];
    out[b] = acc > 0.f ? acc : 0.f;
}

// ---------------------------------------------------------------------------
// Launch
// ---------------------------------------------------------------------------
extern "C" void kernel_launch(void* const* d_in, const int* in_sizes, int n_in,
                              void* d_out, int out_size, void* d_ws, size_t ws_size,
                              hipStream_t stream) {
    const float* dense  = (const float*)d_in[0];
    const int*   sparse = (const int*)  d_in[1];
    const float* emb    = (const float*)d_in[2];
    const float* bw0 = (const float*)d_in[3],  *bb0 = (const float*)d_in[4];
    const float* bw1 = (const float*)d_in[5],  *bb1 = (const float*)d_in[6];
    const float* bw2 = (const float*)d_in[7],  *bb2 = (const float*)d_in[8];
    const float* tw0 = (const float*)d_in[9],  *tb0 = (const float*)d_in[10];
    const float* tw1 = (const float*)d_in[11], *tb1 = (const float*)d_in[12];
    const float* tw2 = (const float*)d_in[13], *tb2 = (const float*)d_in[14];
    const float* tw3 = (const float*)d_in[15], *tb3 = (const float*)d_in[16];
    const float* tw4 = (const float*)d_in[17], *tb4 = (const float*)d_in[18];
    float* out = (float*)d_out;

    char* ws = (char*)d_ws;
    size_t off = 0;
    auto take = [&](size_t bytes) -> char* {
        char* p = ws + off;
        off += (bytes + 255) & ~(size_t)255;
        return p;
    };
    // bf16 weights, TRANSPOSED [N][Kp] (K padded to /32)
    __bf16* w0p = (__bf16*)take((size_t)512  * 32   * 2);
    __bf16* w1b = (__bf16*)take((size_t)256  * 512  * 2);
    __bf16* w2b = (__bf16*)take((size_t)128  * 256  * 2);
    __bf16* t0p = (__bf16*)take((size_t)1024 * 512  * 2);
    __bf16* t1b = (__bf16*)take((size_t)1024 * 1024 * 2);
    __bf16* t2b = (__bf16*)take((size_t)512  * 1024 * 2);
    __bf16* t3b = (__bf16*)take((size_t)256  * 512  * 2);
    // activations
    __bf16* Xd   = (__bf16*)take((size_t)BATCH * 32   * 2);
    __bf16* A1   = (__bf16*)take((size_t)BATCH * 512  * 2);
    __bf16* A2   = (__bf16*)take((size_t)BATCH * 256  * 2);
    __bf16* Dout = (__bf16*)take((size_t)BATCH * 128  * 2);
    __bf16* X0   = (__bf16*)take((size_t)BATCH * 512  * 2);
    __bf16* Feat = (__bf16*)take((size_t)BATCH * 4096 * 2);   // 128 MB
    // X1..X4 alias Feat (dead after interaction)
    __bf16* X1 = Feat;
    __bf16* X2 = (__bf16*)((char*)Feat + (size_t)BATCH * 1024 * 2);
    __bf16* X3 = (__bf16*)((char*)X2   + (size_t)BATCH * 1024 * 2);
    __bf16* X4 = (__bf16*)((char*)X3   + (size_t)BATCH * 512  * 2);

    auto cvt = [&](const float* s, __bf16* d, int K, int N, int Kp) {
        int tot = N * Kp;
        convert_pad_t<<<(tot + 255) / 256, 256, 0, stream>>>(s, d, K, N, Kp);
    };
    cvt(bw0, w0p, 13,   512,  32);
    cvt(bw1, w1b, 512,  256,  512);
    cvt(bw2, w2b, 256,  128,  256);
    cvt(tw0, t0p, 479,  1024, 512);
    cvt(tw1, t1b, 1024, 1024, 1024);
    cvt(tw2, t2b, 1024, 512,  1024);
    cvt(tw3, t3b, 512,  256,  512);
    pad_dense<<<(BATCH * 32 + 255) / 256, 256, 0, stream>>>(dense, Xd);

    auto gemm = [&](const __bf16* A, const __bf16* Wt, const float* bias,
                    __bf16* C, int N, int K) {
        dim3 g(N / 128, BATCH / 128);
        gemm_bf16_relu<<<g, 256, 0, stream>>>(A, Wt, bias, C, BATCH, N, K);
    };
    // bottom MLP
    gemm(Xd, w0p, bb0, A1,   512, 32);
    gemm(A1, w1b, bb1, A2,   256, 512);
    gemm(A2, w2b, bb2, Dout, 128, 256);
    // features + interaction
    gather_feats<<<BATCH, 128, 0, stream>>>(emb, sparse, Dout, Feat, X0);
    interact<<<BATCH / 4, 128, 0, stream>>>(Feat, X0);
    // top MLP
    gemm(X0, t0p, tb0, X1, 1024, 512);
    gemm(X1, t1b, tb1, X2, 1024, 1024);
    gemm(X2, t2b, tb2, X3, 512,  1024);
    gemm(X3, t3b, tb3, X4, 256,  512);
    final_dot<<<BATCH / 256, 256, 0, stream>>>(X4, tw4, tb4, out);
}
